// PredictionLayer_89507118449343
// MI455X (gfx1250) — matrix operand order repaired and checked by reading, not compiled, not run on hardware
//
#include <hip/hip_runtime.h>

#define NUM_ITEMS 50000
#define NUM_USERS 2048
#define EMB 128
#define BATCH 1024
#define SETS 50
#define C_IMP 0.5f

typedef __attribute__((ext_vector_type(8)))  __bf16 v8bf;
typedef __attribute__((ext_vector_type(16))) __bf16 v16bf;
typedef __attribute__((ext_vector_type(8)))  float  v8f;

static __device__ __forceinline__ unsigned short f32_to_bf16_rne(float f) {
    unsigned int u = __float_as_uint(f);
    unsigned int r = (u + 0x7FFFu + ((u >> 16) & 1u)) >> 16;   // round-to-nearest-even
    return (unsigned short)r;
}
static __device__ __forceinline__ float bf16_to_f32(unsigned short h) {
    return __uint_as_float(((unsigned int)h) << 16);
}

// ---------------------------------------------------------------------------
// Kernel 1: U' = U @ W  (stored bf16),  c = U . b
// 64 blocks x 256 thr; 16 batch rows per block; W staged in 64KB LDS.
// ---------------------------------------------------------------------------
__global__ __launch_bounds__(256) void prep_uprime(
    const float* __restrict__ U,      // (B,EMB)
    const float* __restrict__ W,      // (EMB,EMB) row-major: W[e*EMB+k]
    const float* __restrict__ bias,   // (EMB)
    unsigned short* __restrict__ Upb, // (B,EMB) bf16
    float* __restrict__ cvec)         // (B)
{
    __shared__ float Wl[EMB * EMB];   // 64 KB
    const int t = threadIdx.x;
    for (int i = t; i < EMB * EMB; i += 256) Wl[i] = W[i];
    __syncthreads();

    const int row = blockIdx.x * 16 + (t >> 4);
    const int kb  = (t & 15);
    const float* urow = U + row * EMB;

    if (kb == 0) {
        float acc = 0.f;
        #pragma unroll 8
        for (int e = 0; e < EMB; ++e) acc += urow[e] * bias[e];
        cvec[row] = acc;
    }
    #pragma unroll
    for (int j = 0; j < 8; ++j) {
        const int k = kb + 16 * j;
        float acc = 0.f;
        #pragma unroll 8
        for (int e = 0; e < EMB; ++e) acc += urow[e] * Wl[e * EMB + k];
        Upb[row * EMB + k] = f32_to_bf16_rne(acc);
    }
}

// ---------------------------------------------------------------------------
// Kernel 2: items_memory f32 -> bf16 (4 elements / thread)
// ---------------------------------------------------------------------------
__global__ __launch_bounds__(256) void conv_items(
    const float4* __restrict__ src, ushort4* __restrict__ dst, int n4)
{
    int i = blockIdx.x * blockDim.x + threadIdx.x;
    if (i < n4) {
        float4 v = src[i];
        ushort4 o;
        o.x = f32_to_bf16_rne(v.x); o.y = f32_to_bf16_rne(v.y);
        o.z = f32_to_bf16_rne(v.z); o.w = f32_to_bf16_rne(v.w);
        dst[i] = o;
    }
}

// ---------------------------------------------------------------------------
// Kernel 3: fused GEMM + masking.
//   out[m,n] = (1 - 0.5*hist)*p + 0.5*hist*( U'[m,:].items[n,:] + c[m] )
// Workgroup: 256 thr = 8 waves. Block tile: M=16 (blockIdx.y), N=128
// (blockIdx.x); wave w owns 16x16 tile at n0 = bx*128 + w*16.
// K=128 -> 4x v_wmma_f32_16x16x32_bf16, f32 accumulate.
// ---------------------------------------------------------------------------
__global__ __launch_bounds__(256) void pred_main(
    const unsigned short* __restrict__ Upb,    // (B,EMB) bf16
    const unsigned short* __restrict__ itemsb, // (N,EMB) bf16
    const float* __restrict__ cvec,            // (B)
    const int* __restrict__ uid,               // (B)
    const int* __restrict__ hist,              // (NUM_USERS, NUM_ITEMS)
    const float* __restrict__ prob,            // (B, NUM_ITEMS)
    float* __restrict__ outp)                  // (B, NUM_ITEMS)
{
    const int wave  = threadIdx.x >> 5;
    const int lane  = threadIdx.x & 31;
    const int lhalf = lane >> 4;     // 0/1
    const int l16   = lane & 15;

    const int m0 = blockIdx.y * 16;
    const int n0 = blockIdx.x * 128 + wave * 16;

    // A (16x32 bf16) per-lane layout: row = l16; lanes 0-15 hold K0..15,
    // lanes 16-31 hold K16..31  -> two contiguous 8-elem chunks.
    const unsigned short* arow = Upb + (m0 + l16) * EMB + lhalf * 16;

    // B (32x16 bf16): col = l16; lanes 0-15 hold K0..15, lanes 16-31 K16..31
    // -> one contiguous 16-elem (32B) chunk of the item's row.
    const int  nitem  = n0 + l16;
    const int  nclamp = nitem < NUM_ITEMS ? nitem : (NUM_ITEMS - 1);
    const unsigned short* brow = itemsb + (size_t)nclamp * EMB + lhalf * 16;

    v8f acc = {};
    #pragma unroll
    for (int kc = 0; kc < EMB; kc += 32) {
        v8bf a_lo = *(const v8bf*)(arow + kc);
        v8bf a_hi = *(const v8bf*)(arow + kc + 8);
        v16bf af  = __builtin_shufflevector(a_lo, a_hi,
                        0,1,2,3,4,5,6,7,8,9,10,11,12,13,14,15);
        v16bf bfr = *(const v16bf*)(brow + kc);   // 32B aligned
        acc = __builtin_amdgcn_wmma_f32_16x16x32_bf16(
                  false, af, false, bfr, (short)0, acc, false, false);
    }

    // C/D layout: VGPR g -> row m0 + g + 8*lhalf, col = n0 + l16.
    if (nitem < NUM_ITEMS) {
        #pragma unroll
        for (int g = 0; g < 8; ++g) {
            const int m = m0 + g + lhalf * 8;
            const int u = uid[m];
            const float h = (hist[(size_t)u * NUM_ITEMS + nitem] != 0) ? 1.0f : 0.0f;
            const float p = prob[(size_t)m * NUM_ITEMS + nitem];
            const float hp = C_IMP * (acc[g] + cvec[m]);
            outp[(size_t)m * NUM_ITEMS + nitem] = (1.0f - C_IMP * h) * p + hp * h;
        }
    }
}

// ---------------------------------------------------------------------------
// Kernel 4: in-set fixup.  For valid (b,s), at n = ids[b,s]:
//   out += 0.5*ct - 0.5*hist*(U'[b].items[n] + c[b])
// (removes the history contribution at set positions and adds the ct term)
// ---------------------------------------------------------------------------
__global__ __launch_bounds__(256) void pred_fixup(
    const int* __restrict__ blen,
    const int* __restrict__ ids,              // (B,S)
    const int* __restrict__ uid,
    const int* __restrict__ hist,
    const float* __restrict__ umem,           // (B,EMB)
    const float* __restrict__ imem,           // (B,S,EMB)
    const unsigned short* __restrict__ Upb,
    const unsigned short* __restrict__ itemsb,
    const float* __restrict__ cvec,
    float* __restrict__ outp)
{
    const int t = blockIdx.x * blockDim.x + threadIdx.x;
    if (t >= BATCH * SETS) return;
    const int b = t / SETS, s = t % SETS;
    if (s >= blen[b]) return;
    const int n = ids[b * SETS + s];

    const float* ur = umem + b * EMB;
    const float* ir = imem + ((size_t)b * SETS + s) * EMB;
    float ct = 0.f;
    #pragma unroll 8
    for (int e = 0; e < EMB; ++e) ct += ur[e] * ir[e];

    const float h = (hist[(size_t)uid[b] * NUM_ITEMS + n] != 0) ? 1.f : 0.f;

    const unsigned short* a  = Upb + b * EMB;
    const unsigned short* bb = itemsb + (size_t)n * EMB;
    float hp = 0.f;
    #pragma unroll 8
    for (int e = 0; e < EMB; ++e) hp += bf16_to_f32(a[e]) * bf16_to_f32(bb[e]);
    hp = C_IMP * (hp + cvec[b]);

    outp[(size_t)b * NUM_ITEMS + n] += C_IMP * ct - hp * h;
}

// ---------------------------------------------------------------------------
extern "C" void kernel_launch(void* const* d_in, const int* in_sizes, int n_in,
                              void* d_out, int out_size, void* d_ws, size_t ws_size,
                              hipStream_t stream) {
    const float* items_memory = (const float*)d_in[0];
    const int*   batch_length = (const int*)d_in[1];
    const int*   batch_user_id = (const int*)d_in[2];
    const int*   batch_items_id = (const int*)d_in[3];
    const int*   users_history_items = (const int*)d_in[4];
    const float* batch_user_memory = (const float*)d_in[5];
    const float* batch_items_memory = (const float*)d_in[6];
    const float* prob = (const float*)d_in[7];
    const float* W = (const float*)d_in[8];
    const float* bvec = (const float*)d_in[9];
    float* outp = (float*)d_out;

    // Workspace layout (needs ~13.1 MB):
    //   [0,      262144)  U' bf16  (1024x128)
    //   [262144, 266240)  c        (1024 f32)
    //   [266240, ...   )  items bf16 (50000x128), 256B-aligned base
    char* ws = (char*)d_ws;
    unsigned short* Upb   = (unsigned short*)(ws);
    float*          cvec  = (float*)(ws + 262144);
    unsigned short* itemsb = (unsigned short*)(ws + 266240);

    prep_uprime<<<BATCH / 16, 256, 0, stream>>>(
        batch_user_memory, W, bvec, Upb, cvec);

    const int n4 = NUM_ITEMS * EMB / 4;
    conv_items<<<(n4 + 255) / 256, 256, 0, stream>>>(
        (const float4*)items_memory, (ushort4*)itemsb, n4);

    dim3 grid((NUM_ITEMS + 127) / 128, BATCH / 16);
    pred_main<<<grid, 256, 0, stream>>>(
        Upb, itemsb, cvec, batch_user_id, users_history_items, prob, outp);

    pred_fixup<<<(BATCH * SETS + 255) / 256, 256, 0, stream>>>(
        batch_length, batch_items_id, batch_user_id, users_history_items,
        batch_user_memory, batch_items_memory, Upb, itemsb, cvec, outp);
}
